// CausalSelfAttention_32624571580905
// MI455X (gfx1250) — matrix operand order repaired and checked by reading, not compiled, not run on hardware
//
#include <hip/hip_runtime.h>

// ---------------------------------------------------------------------------
// Causal self-attention forward for gfx1250 (MI455X), wave32 + WMMA bf16
// + async global->LDS DMA (GLOBAL_LOAD_ASYNC_TO_LDS_B128) with LDS double
// buffering so DMA of tile j+1 overlaps WMMA on tile j.
// Pipeline:
//   1) convert x -> bf16, transpose+convert Wq/Wk/Wv/Wp -> bf16 [N][K]
//   2) GEMM (WMMA bf16): q,k -> [B,H,T,D] bf16 ; v -> [B,H,D,T] bf16
//   3) flash attention (WMMA bf16, online softmax, causal tile skipping)
//      -> y bf16 [B,T,C]
//   4) GEMM (WMMA bf16): out = y @ Wp + bp -> f32 [B,T,C]
// ---------------------------------------------------------------------------

typedef __bf16 bf16_t;
typedef __attribute__((ext_vector_type(16))) __bf16        v16bf;
typedef __attribute__((ext_vector_type(8)))  float         v8f;
typedef __attribute__((ext_vector_type(4)))  unsigned int  u32x4;
typedef int g4i __attribute__((vector_size(16)));  // matches builtin param

// ---- async global -> LDS (gfx1250), with portable fallback -----------------
#if defined(__HIP_DEVICE_COMPILE__) && \
    __has_builtin(__builtin_amdgcn_global_load_async_to_lds_b128)
#define ASYNC_LDS 1
__device__ __forceinline__ void gld_async16(const bf16_t* g, bf16_t* l) {
  __builtin_amdgcn_global_load_async_to_lds_b128(
      (__attribute__((address_space(1))) g4i*)g,
      (__attribute__((address_space(3))) g4i*)l, 0, 0);
}
__device__ __forceinline__ void wait_async() {
#if __has_builtin(__builtin_amdgcn_s_wait_asynccnt)
  __builtin_amdgcn_s_wait_asynccnt(0);
#else
  asm volatile("s_wait_asynccnt 0" ::: "memory");
#endif
}
#else
#define ASYNC_LDS 0
__device__ __forceinline__ void gld_async16(const bf16_t* g, bf16_t* l) {
  *(u32x4*)l = *(const u32x4*)g;  // global_load_b128 + ds_store_b128
}
__device__ __forceinline__ void wait_async() {}
#endif

#define B_  4
#define T_  2048
#define C_  768
#define H_  12
#define D_  64
#define M_  (B_ * T_)   // 8192

// ---- WMMA fragment loaders (layouts per CDNA5 ISA 7.12.2) ------------------
// A-matrix 16x32 bf16, row m in lane m%16, half h=lane/16:
//   VGPR0-3: K = h*8 + 0..7 ; VGPR4-7: K = 16 + h*8 + 0..7
__device__ __forceinline__ v16bf load_fragA32(const bf16_t* kwin, int hf) {
  union { u32x4 u[2]; v16bf v; } t;
  const u32x4* p = (const u32x4*)kwin;
  t.u[0] = p[hf];
  t.u[1] = p[2 + hf];
  return t.v;
}
// B-matrix 32x16 bf16, col n in lane n%16, half h=lane/16:
//   VGPR0-3: K = h*8 + 0..7 ; VGPR4-7: K = 16 + h*8 + 0..7
__device__ __forceinline__ v16bf load_fragB32(const bf16_t* kwin, int hf) {
  union { u32x4 u[2]; v16bf v; } t;
  const u32x4* p = (const u32x4*)kwin;
  t.u[0] = p[hf];
  t.u[1] = p[2 + hf];
  return t.v;
}
__device__ __forceinline__ v8f wmma_bf16(v16bf a, v16bf b, v8f c) {
  return __builtin_amdgcn_wmma_f32_16x16x32_bf16(
      false, a, false, b, (short)0, c, false, false);
}

// ---- conversion kernels ----------------------------------------------------
__global__ void cvt_f32_bf16(const float* __restrict__ src,
                             bf16_t* __restrict__ dst, int n) {
  int i = blockIdx.x * blockDim.x + threadIdx.x;
  if (i < n) dst[i] = (bf16_t)src[i];
}

// W[k][n] f32 (C_ x C_) -> Wt[n][k] bf16
__global__ void transpose_w(const float* __restrict__ W,
                            bf16_t* __restrict__ Wt) {
  int i = blockIdx.x * blockDim.x + threadIdx.x;  // over C_*C_
  int n = i / C_, k = i % C_;
  Wt[i] = (bf16_t)W[k * C_ + n];
}

// ---- GEMM: A[M][K] bf16 x Bt[N][K] bf16 + bias -> Out ----------------------
// Block tile 128x128, 8 waves, each wave owns a 32x64 sub-tile (8 WMMA/step).
// Double-buffered LDS, async DMA of tile j+1 overlapped with WMMA on tile j.
// mode 0: bf16 [B,H,T,D] ; mode 1: bf16 [B,H,D,T] ; mode 2: f32 [M][N]
#define GM 128
#define GN 128
#define GK 32

__global__ __launch_bounds__(256) void gemm_bf16(
    const bf16_t* __restrict__ A, const bf16_t* __restrict__ Bt,
    const float* __restrict__ bias, void* __restrict__ Out,
    int M, int N, int K, int mode) {
  __shared__ __attribute__((aligned(16))) bf16_t As[2][GM][GK];  // 16 KB
  __shared__ __attribute__((aligned(16))) bf16_t Bs[2][GN][GK];  // 16 KB

  const int tid  = threadIdx.x;
  const int wave = tid >> 5, lane = tid & 31;
  const int ln16 = lane & 15, hf = lane >> 4;
  const int wm = wave >> 1, wn = wave & 1;   // 4 x 2 wave grid
  const int m0 = blockIdx.x * GM;
  const int n0 = blockIdx.y * GN;

  const int lrow = tid >> 1, loff = (tid & 1) * 16;  // 2x16B per array/thread
  const bf16_t* gA = A  + (size_t)(m0 + lrow) * K + loff;
  const bf16_t* gB = Bt + (size_t)(n0 + lrow) * K + loff;

  v8f acc[2][4] = {};

  // prologue: tile 0 -> buffer 0
  gld_async16(gA,     &As[0][lrow][loff]);
  gld_async16(gA + 8, &As[0][lrow][loff + 8]);
  gld_async16(gB,     &Bs[0][lrow][loff]);
  gld_async16(gB + 8, &Bs[0][lrow][loff + 8]);
  wait_async();
  __syncthreads();

  int p = 0;
  for (int k0 = 0; k0 < K; k0 += GK, p ^= 1) {
    if (k0 + GK < K) {  // DMA next tile while computing this one
      const bf16_t* nA = gA + (k0 + GK);
      const bf16_t* nB = gB + (k0 + GK);
      gld_async16(nA,     &As[p ^ 1][lrow][loff]);
      gld_async16(nA + 8, &As[p ^ 1][lrow][loff + 8]);
      gld_async16(nB,     &Bs[p ^ 1][lrow][loff]);
      gld_async16(nB + 8, &Bs[p ^ 1][lrow][loff + 8]);
    }
    v16bf af0 = load_fragA32(&As[p][wm * 32 + ln16][0], hf);
    v16bf af1 = load_fragA32(&As[p][wm * 32 + 16 + ln16][0], hf);
#pragma unroll
    for (int nt = 0; nt < 4; ++nt) {
      v16bf bfr = load_fragB32(&Bs[p][wn * 64 + nt * 16 + ln16][0], hf);
      acc[0][nt] = wmma_bf16(af0, bfr, acc[0][nt]);
      acc[1][nt] = wmma_bf16(af1, bfr, acc[1][nt]);
    }
    if (k0 + GK < K) wait_async();
    __syncthreads();
  }

  // D layout: lane holds column n=ln16; rows m = 8*hf + r
#pragma unroll
  for (int mt = 0; mt < 2; ++mt) {
    const int mbase = m0 + wm * 32 + mt * 16 + hf * 8;
#pragma unroll
    for (int nt = 0; nt < 4; ++nt) {
      const int n = n0 + wn * 64 + nt * 16 + ln16;
      const float bv = bias[n];
      if (mode == 2) {
        float* O = (float*)Out;
#pragma unroll
        for (int r = 0; r < 8; ++r)
          O[(size_t)(mbase + r) * N + n] = acc[mt][nt][r] + bv;
      } else {
        bf16_t* O = (bf16_t*)Out;
        const int h = n >> 6, d = n & 63;
#pragma unroll
        for (int r = 0; r < 8; ++r) {
          const int m = mbase + r;
          const int bb = m >> 11, t = m & (T_ - 1);
          size_t addr;
          if (mode == 0) addr = (((size_t)bb * H_ + h) * T_ + t) * D_ + d;
          else           addr = (((size_t)bb * H_ + h) * D_ + d) * T_ + t;
          O[addr] = (bf16_t)(acc[mt][nt][r] + bv);
        }
      }
    }
  }
}

// ---- flash attention: 64 q-rows/block, 64-key tiles, causal, double-buffered
__global__ __launch_bounds__(128) void flash_attn(
    const bf16_t* __restrict__ Q,   // [B,H,T,D]
    const bf16_t* __restrict__ Km,  // [B,H,T,D]
    const bf16_t* __restrict__ Vt,  // [B,H,D,T]
    bf16_t* __restrict__ Y) {       // [B,T,C]
  __shared__ __attribute__((aligned(16))) bf16_t Qs[64][64];        // 8 KB
  __shared__ __attribute__((aligned(16))) bf16_t Ks[2][64][64];     // 16 KB
  __shared__ __attribute__((aligned(16))) bf16_t Vs[2][64][64];     // 16 KB
  __shared__ __attribute__((aligned(16))) bf16_t Ps[4][16][64];     // 8 KB

  const int tid  = threadIdx.x;
  const int wave = tid >> 5, lane = tid & 31;
  const int ln16 = lane & 15, hf = lane >> 4;
  const int bh = blockIdx.y;               // b*H + h
  const int b  = bh / H_, h = bh % H_;
  const int q0 = blockIdx.x * 64;
  const size_t head = (size_t)bh * T_ * D_;

  const int lrow = tid >> 1, loff = (tid & 1) * 32;  // 64B per array/thread

  auto issue_kv = [&](int kt, int pb) {
    const bf16_t* sk = Km + head + (size_t)(kt * 64 + lrow) * D_ + loff;
    const bf16_t* sv = Vt + head + (size_t)lrow * T_ + kt * 64 + loff;
#pragma unroll
    for (int i = 0; i < 4; ++i) {
      gld_async16(sk + i * 8, &Ks[pb][lrow][loff + i * 8]);
      gld_async16(sv + i * 8, &Vs[pb][lrow][loff + i * 8]);
    }
  };

  // prologue: Q tile + first K/V tile via async DMA
  {
    const bf16_t* sq = Q + head + (size_t)(q0 + lrow) * D_ + loff;
#pragma unroll
    for (int i = 0; i < 4; ++i)
      gld_async16(sq + i * 8, &Qs[lrow][loff + i * 8]);
  }
  issue_kv(0, 0);
  wait_async();
  __syncthreads();

  v16bf qf0, qf1;
  {
    const bf16_t* qrow = &Qs[wave * 16 + ln16][0];
    qf0 = load_fragA32(qrow, hf);        // d 0..31
    qf1 = load_fragA32(qrow + 32, hf);   // d 32..63
  }

  v8f   accO[4] = {};
  float mrow[8], lrow_[8];
#pragma unroll
  for (int r = 0; r < 8; ++r) { mrow[r] = -__builtin_inff(); lrow_[r] = 0.f; }

  const float scale = 0.125f;  // 1/sqrt(64)
  const int qbase  = q0 + wave * 16 + hf * 8;
  const int ntiles = ((q0 + 63) >> 6) + 1;  // causal: skip past-diagonal tiles

  for (int kt = 0; kt < ntiles; ++kt) {
    const int p  = kt & 1;
    const int k0 = kt * 64;
    if (kt + 1 < ntiles) issue_kv(kt + 1, p ^ 1);  // overlap DMA with compute

    // S = q k^T : 4 key subtiles x (2 K-halves of D)
    v8f s[4];
#pragma unroll
    for (int nt = 0; nt < 4; ++nt) {
      const bf16_t* krow = &Ks[p][nt * 16 + ln16][0];
      v8f z = {};
      z = wmma_bf16(qf0, load_fragB32(krow, hf), z);
      z = wmma_bf16(qf1, load_fragB32(krow + 32, hf), z);
      s[nt] = z;
    }

    // scale + causal mask (lane holds key column nt*16+ln16; rows qbase+r)
#pragma unroll
    for (int nt = 0; nt < 4; ++nt) {
      const int key = k0 + nt * 16 + ln16;
#pragma unroll
      for (int r = 0; r < 8; ++r) {
        float v = s[nt][r] * scale;
        s[nt][r] = (key <= qbase + r) ? v : -__builtin_inff();
      }
    }

    // online softmax: row stats via 16-lane xor-shuffle reductions
#pragma unroll
    for (int r = 0; r < 8; ++r) {
      float mx = fmaxf(fmaxf(s[0][r], s[1][r]), fmaxf(s[2][r], s[3][r]));
#pragma unroll
      for (int sm = 1; sm < 16; sm <<= 1)
        mx = fmaxf(mx, __shfl_xor(mx, sm, 32));
      const float mnew  = fmaxf(mrow[r], mx);
      const float alpha = __expf(mrow[r] - mnew);
      float rsum = 0.f;
#pragma unroll
      for (int nt = 0; nt < 4; ++nt) {
        float pv = __expf(s[nt][r] - mnew);
        s[nt][r] = pv;
        rsum += pv;
      }
#pragma unroll
      for (int sm = 1; sm < 16; sm <<= 1)
        rsum += __shfl_xor(rsum, sm, 32);
      lrow_[r] = lrow_[r] * alpha + rsum;
      mrow[r] = mnew;
#pragma unroll
      for (int dt = 0; dt < 4; ++dt) accO[dt][r] *= alpha;
    }

    // P (C-layout) -> wave-private LDS -> reload in A-layout (same-wave LDS
    // ops are in-order; compiler inserts s_wait_dscnt, no barrier needed)
#pragma unroll
    for (int nt = 0; nt < 4; ++nt)
#pragma unroll
      for (int r = 0; r < 8; ++r)
        Ps[wave][hf * 8 + r][nt * 16 + ln16] = (bf16_t)s[nt][r];

    const bf16_t* prow = &Ps[wave][ln16][0];
    v16bf pf0 = load_fragA32(prow, hf);        // keys 0..31
    v16bf pf1 = load_fragA32(prow + 32, hf);   // keys 32..63

    // O += P V : B-operand from Vt tile [d][key]
#pragma unroll
    for (int dt = 0; dt < 4; ++dt) {
      const bf16_t* vrow = &Vs[p][dt * 16 + ln16][0];
      accO[dt] = wmma_bf16(pf0, load_fragB32(vrow, hf), accO[dt]);
      accO[dt] = wmma_bf16(pf1, load_fragB32(vrow + 32, hf), accO[dt]);
    }

    if (kt + 1 < ntiles) wait_async();
    __syncthreads();
  }

  // epilogue: O /= l ; write y[B,T,C] with c = h*64 + d
#pragma unroll
  for (int dt = 0; dt < 4; ++dt) {
    const int d = dt * 16 + ln16;
#pragma unroll
    for (int r = 0; r < 8; ++r) {
      const int t = qbase + r;
      const float o = accO[dt][r] / lrow_[r];
      Y[((size_t)b * T_ + t) * C_ + h * D_ + d] = (bf16_t)o;
    }
  }
}

// ---------------------------------------------------------------------------
extern "C" void kernel_launch(void* const* d_in, const int* in_sizes, int n_in,
                              void* d_out, int out_size, void* d_ws,
                              size_t ws_size, hipStream_t stream) {
  (void)in_sizes; (void)n_in; (void)out_size; (void)ws_size;
  const float* x  = (const float*)d_in[0];
  const float* Wq = (const float*)d_in[1];
  const float* bq = (const float*)d_in[2];
  const float* Wk = (const float*)d_in[3];
  const float* bk = (const float*)d_in[4];
  const float* Wv = (const float*)d_in[5];
  const float* bv = (const float*)d_in[6];
  const float* Wp = (const float*)d_in[7];
  const float* bp = (const float*)d_in[8];

  // workspace layout (bf16)
  char* ws = (char*)d_ws;
  const size_t xN = (size_t)M_ * C_;        // 6291456
  const size_t wN = (size_t)C_ * C_;        // 589824
  bf16_t* xb  = (bf16_t*)(ws);                              // x bf16
  bf16_t* Wqt = (bf16_t*)(ws + 2 * xN);                     // Wt's
  bf16_t* Wkt = Wqt + wN;
  bf16_t* Wvt = Wkt + wN;
  bf16_t* Wpt = Wvt + wN;
  bf16_t* q   = (bf16_t*)((char*)(Wpt + wN));               // [B,H,T,D]
  bf16_t* k   = q + xN;                                     // [B,H,T,D]
  bf16_t* vT  = k + xN;                                     // [B,H,D,T]
  bf16_t* y   = vT + xN;                                    // [B,T,C]

  // 1) conversions
  cvt_f32_bf16<<<(int)((xN + 255) / 256), 256, 0, stream>>>(x, xb, (int)xN);
  transpose_w<<<(int)(wN / 256), 256, 0, stream>>>(Wq, Wqt);
  transpose_w<<<(int)(wN / 256), 256, 0, stream>>>(Wk, Wkt);
  transpose_w<<<(int)(wN / 256), 256, 0, stream>>>(Wv, Wvt);
  transpose_w<<<(int)(wN / 256), 256, 0, stream>>>(Wp, Wpt);

  // 2) QKV projections
  dim3 ggrid(M_ / GM, C_ / GN);  // 64 x 6
  gemm_bf16<<<ggrid, 256, 0, stream>>>(xb, Wqt, bq, q,  M_, C_, C_, 0);
  gemm_bf16<<<ggrid, 256, 0, stream>>>(xb, Wkt, bk, k,  M_, C_, C_, 0);
  gemm_bf16<<<ggrid, 256, 0, stream>>>(xb, Wvt, bv, vT, M_, C_, C_, 1);

  // 3) flash attention
  dim3 agrid(T_ / 64, B_ * H_);  // 32 x 48
  flash_attn<<<agrid, 128, 0, stream>>>(q, k, vT, y);

  // 4) output projection -> f32
  gemm_bf16<<<ggrid, 256, 0, stream>>>(y, Wpt, bp, (float*)d_out,
                                       M_, C_, C_, 2);
}
